// HGAT_22411139350654
// MI455X (gfx1250) — matrix operand order repaired and checked by reading, not compiled, not run on hardware
//
#include <hip/hip_runtime.h>
#include <cmath>

// ---------------------------------------------------------------------------
// CDNA5 (gfx1250) HGAT pipeline.  wave32.  WMMA bf16 16x16x32 for all GEMMs.
// ---------------------------------------------------------------------------

typedef __bf16 bf16;
typedef __attribute__((ext_vector_type(16))) __bf16 v16bf;
typedef __attribute__((ext_vector_type(8)))  float  v8f;

#define KN 32   // neighbors per node (fixed in reference)
#define HH 4    // heads
#define DD 64   // feature dim

// ---------------------------------------------------------------------------
// uv_kernel: v[rel][h][c] = sum_o Ws[h][c][o]*a[h][0][o]
//            u[rel][h][c] = sum_o Wn[h][c][o]*a[h][1][o]
// layout uv[rel][which(0=v,1=u)][h][c]   (4*2*4*64 floats)
// ---------------------------------------------------------------------------
__global__ void uv_kernel(const float* __restrict__ Ws0, const float* __restrict__ Wn0, const float* __restrict__ a0,
                          const float* __restrict__ Ws1, const float* __restrict__ Wn1, const float* __restrict__ a1,
                          const float* __restrict__ Ws2, const float* __restrict__ Wn2, const float* __restrict__ a2,
                          const float* __restrict__ Ws3, const float* __restrict__ Wn3, const float* __restrict__ a3,
                          float* __restrict__ uv)
{
    int tid = blockIdx.x * blockDim.x + threadIdx.x;
    if (tid >= 4 * 2 * HH * DD) return;
    int c     = tid & 63;
    int h     = (tid >> 6) & 3;
    int which = (tid >> 8) & 1;
    int rel   = tid >> 9;
    const float *Ws, *Wn, *a;
    if      (rel == 0) { Ws = Ws0; Wn = Wn0; a = a0; }
    else if (rel == 1) { Ws = Ws1; Wn = Wn1; a = a1; }
    else if (rel == 2) { Ws = Ws2; Wn = Wn2; a = a2; }
    else               { Ws = Ws3; Wn = Wn3; a = a3; }
    const float* M  = which ? Wn : Ws;                 // [H][64][64]
    const float* av = a + h * 128 + which * 64;        // a[h][which][:]
    const float* Mr = M + h * 4096 + c * 64;
    float acc = 0.f;
    for (int o = 0; o < DD; ++o) acc += Mr[o] * av[o];
    uv[tid] = acc;
}

// ---------------------------------------------------------------------------
// wfrag_kernel: convert W [Kdim][64] (f32 row-major) into bf16 WMMA-B operand
// fragments: out[ks][nt][lane][e], col = nt*16+(lane&15), k = ks*32+(e&7)+((e>>3)<<4)+(lane>>4)*8
// ---------------------------------------------------------------------------
__global__ void wfrag_kernel(const float* __restrict__ W, bf16* __restrict__ out, int Kdim)
{
    int tid = blockIdx.x * blockDim.x + threadIdx.x;
    int total = (Kdim / 32) * 4 * 32 * 16;
    if (tid >= total) return;
    int e    = tid & 15;
    int lane = (tid >> 4) & 31;
    int nt   = (tid >> 9) & 3;
    int ks   = tid >> 11;
    int col  = nt * 16 + (lane & 15);
    int k    = ks * 32 + (e & 7) + ((e >> 3) << 4) + (lane >> 4) * 8;
    out[tid] = (bf16)W[(size_t)k * 64 + col];
}

// ---------------------------------------------------------------------------
// node_attn_kernel: one wave handles a 16-node tile.
//   scores: s[n,h,k] = leaky(hc[n].v_h + nb[n,k].u_h); alpha = softmax_k
//   m[n,h,:] = sum_k alpha*nb[n,k,:]
//   t[n,h,:] = elu(m[n,h,:] @ Wn[h])            (WMMA)
//   out[n,:] = t[n,:] @ Wout + b_out            (WMMA)
// ---------------------------------------------------------------------------
__global__ __launch_bounds__(32)
void node_attn_kernel(const float* __restrict__ hc_emb,
                      const int*   __restrict__ center_idx,   // may be null -> identity
                      const float* __restrict__ nb_emb,
                      const int*   __restrict__ idx,          // [*][KN]
                      const float* __restrict__ v_vec,        // [H][64]
                      const float* __restrict__ u_vec,        // [H][64]
                      const bf16*  __restrict__ WnF,          // [8][4][32][16] bf16 frags
                      const bf16*  __restrict__ WoF,          // [8][4][32][16] bf16 frags
                      const float* __restrict__ b_out,        // [64]
                      float* __restrict__ out_f32,            // [N][64]
                      bf16*  __restrict__ out_bf16,           // [N][64] or null
                      int N)
{
    __shared__ bf16  nbS[8][KN][DD];      // 32 KB (half tile of neighbor features)
    __shared__ float sc [8][HH][KN];      //  4 KB (scores -> alpha)
    __shared__ float cS [8][HH];
    __shared__ bf16  mS [16][HH * DD];    //  8 KB (A operand for t-GEMM)
    __shared__ bf16  tS [16][HH * DD];    //  8 KB (A operand for out-GEMM)

    const int lane  = threadIdx.x;
    const int node0 = blockIdx.x * 16;

    for (int half = 0; half < 2; ++half) {
        const int nbase = node0 + half * 8;
        // --- gather neighbor embeddings -> LDS (bf16) ---
        for (int p = lane; p < 8 * KN; p += 32) {
            int nl = p >> 5;
            int k  = p & 31;
            int n  = nbase + nl; if (n >= N) n = N - 1;
            int cid = center_idx ? center_idx[n] : n;
            int nr  = idx[(size_t)cid * KN + k];
            const float4* src = (const float4*)(nb_emb + (size_t)nr * DD);
            #pragma unroll
            for (int q = 0; q < 16; ++q) {
                float4 f = src[q];
                nbS[nl][k][q * 4 + 0] = (bf16)f.x;
                nbS[nl][k][q * 4 + 1] = (bf16)f.y;
                nbS[nl][k][q * 4 + 2] = (bf16)f.z;
                nbS[nl][k][q * 4 + 3] = (bf16)f.w;
            }
        }
        // --- center score c[n,h] = hc[n].v_h  (one dot per lane) ---
        {
            int nl = lane >> 2, h = lane & 3;
            int n = nbase + nl; if (n >= N) n = N - 1;
            int cid = center_idx ? center_idx[n] : n;
            const float* hc = hc_emb + (size_t)cid * DD;
            const float* vv = v_vec + h * DD;
            float acc = 0.f;
            for (int dd = 0; dd < DD; ++dd) acc += hc[dd] * vv[dd];
            cS[nl][h] = acc;
        }
        __syncthreads();
        // --- neighbor scores + leaky relu ---
        for (int p = lane; p < 8 * HH * KN; p += 32) {
            int k = p & 31, h = (p >> 5) & 3, nl = p >> 7;
            const float* uu = u_vec + h * DD;
            float acc = cS[nl][h];
            for (int dd = 0; dd < DD; ++dd) acc += (float)nbS[nl][k][dd] * uu[dd];
            sc[nl][h][k] = acc > 0.f ? acc : 0.2f * acc;
        }
        __syncthreads();
        // --- softmax over k (one (nl,h) row per lane) ---
        {
            int nl = lane >> 2, h = lane & 3;
            float mx = -1e30f;
            for (int k = 0; k < KN; ++k) mx = fmaxf(mx, sc[nl][h][k]);
            float s = 0.f;
            for (int k = 0; k < KN; ++k) { float e = __expf(sc[nl][h][k] - mx); sc[nl][h][k] = e; s += e; }
            float inv = 1.f / s;
            for (int k = 0; k < KN; ++k) sc[nl][h][k] *= inv;
        }
        __syncthreads();
        // --- m[n,h,:] = sum_k alpha * nb ---
        for (int p = lane; p < 8 * HH * DD; p += 32) {
            int dd = p & 63, h = (p >> 6) & 3, nl = p >> 8;
            float acc = 0.f;
            for (int k = 0; k < KN; ++k) acc += sc[nl][h][k] * (float)nbS[nl][k][dd];
            mS[half * 8 + nl][h * DD + dd] = (bf16)acc;
        }
        __syncthreads();
    }

    const int row = lane & 15;   // also the N-column within a 16-wide tile
    const int hi  = lane >> 4;

    // --- t = elu(m @ Wn[h]) : per head, 16x64 @ 64x64 -> 4 nt * 2 ks WMMAs ---
    for (int h = 0; h < HH; ++h) {
        for (int nt = 0; nt < 4; ++nt) {
            v8f acc = {};
            #pragma unroll
            for (int ks = 0; ks < 2; ++ks) {
                v16bf A, Bv;
                #pragma unroll
                for (int e = 0; e < 16; ++e) {
                    int kl = (e & 7) + ((e >> 3) << 4) + hi * 8;     // A layout K index
                    A[e] = mS[row][h * DD + ks * 32 + kl];
                }
                Bv = *(const v16bf*)(WnF + ((((h * 2 + ks) * 4 + nt) * 32) + lane) * 16);
                acc = __builtin_amdgcn_wmma_f32_16x16x32_bf16(false, A, false, Bv,
                                                              (short)0, acc, false, false);
            }
            #pragma unroll
            for (int j = 0; j < 8; ++j) {
                float x = acc[j];
                x = x > 0.f ? x : (__expf(x) - 1.f);                 // ELU
                tS[j + 8 * hi][h * DD + nt * 16 + row] = (bf16)x;
            }
        }
    }
    __syncthreads();

    // --- out = t @ Wout + b_out : 16x256 @ 256x64 -> 4 nt * 8 ks WMMAs ---
    for (int nt = 0; nt < 4; ++nt) {
        v8f acc = {};
        #pragma unroll
        for (int ks = 0; ks < 8; ++ks) {
            v16bf A, Bv;
            #pragma unroll
            for (int e = 0; e < 16; ++e) {
                int kl = (e & 7) + ((e >> 3) << 4) + hi * 8;
                A[e] = tS[row][ks * 32 + kl];
            }
            Bv = *(const v16bf*)(WoF + (((ks * 4 + nt) * 32) + lane) * 16);
            acc = __builtin_amdgcn_wmma_f32_16x16x32_bf16(false, A, false, Bv,
                                                          (short)0, acc, false, false);
        }
        float bias = b_out[nt * 16 + row];
        #pragma unroll
        for (int j = 0; j < 8; ++j) {
            int n = node0 + j + 8 * hi;
            if (n < N) {
                float r = acc[j] + bias;
                out_f32[(size_t)n * DD + nt * 16 + row] = r;
                if (out_bf16) out_bf16[(size_t)n * DD + nt * 16 + row] = (bf16)r;
            }
        }
    }
}

// ---------------------------------------------------------------------------
// relattn_kernel: score[r] = q . tanh(S_r @ W_rel + b_rel); beta = softmax_r;
// H_R = sum_r beta * S_r  (stored bf16 for the pred GEMM)
// ---------------------------------------------------------------------------
__global__ void relattn_kernel(const float* __restrict__ hru,
                               const float* __restrict__ hrr,
                               const float* __restrict__ hri,
                               const float* __restrict__ W_rel,
                               const float* __restrict__ b_rel,
                               const float* __restrict__ q_rel,
                               bf16* __restrict__ HRbf, int N)
{
    int n = blockIdx.x * blockDim.x + threadIdx.x;
    if (n >= N) return;
    const float* S0 = hru + (size_t)n * DD;
    const float* S1 = hrr + (size_t)n * DD;
    const float* S2 = hri + (size_t)n * DD;
    float score[3];
    #pragma unroll
    for (int r = 0; r < 3; ++r) {
        const float* S = (r == 0) ? S0 : (r == 1) ? S1 : S2;
        float t[DD];
        for (int o = 0; o < DD; ++o) t[o] = b_rel[o];
        for (int c = 0; c < DD; ++c) {
            float sv = S[c];
            const float* wr = W_rel + c * DD;
            for (int o = 0; o < DD; ++o) t[o] += sv * wr[o];
        }
        float acc = 0.f;
        for (int o = 0; o < DD; ++o) acc += tanhf(t[o]) * q_rel[o];
        score[r] = acc;
    }
    float mx = fmaxf(score[0], fmaxf(score[1], score[2]));
    float e0 = __expf(score[0] - mx), e1 = __expf(score[1] - mx), e2 = __expf(score[2] - mx);
    float inv = 1.f / (e0 + e1 + e2);
    e0 *= inv; e1 *= inv; e2 *= inv;
    for (int c = 0; c < DD; ++c) {
        float hv = e0 * S0[c] + e1 * S1[c] + e2 * S2[c];
        HRbf[(size_t)n * DD + c] = (bf16)hv;
    }
}

// ---------------------------------------------------------------------------
// pred_kernel: pred[b,n] = H_U[b,:].H_R[n,:]  -> WMMA 16x16 tiles, K=64
// ---------------------------------------------------------------------------
__global__ __launch_bounds__(32)
void pred_kernel(const bf16* __restrict__ HU, const bf16* __restrict__ HR,
                 float* __restrict__ out, int B, int Nn)
{
    const int lane = threadIdx.x;
    const int row  = lane & 15;
    const int hi   = lane >> 4;
    const int bt   = blockIdx.y * 16;
    const int nt   = blockIdx.x * 16;
    int br = bt + row; if (br >= B)  br = B - 1;
    int nc = nt + row; if (nc >= Nn) nc = Nn - 1;
    v8f acc = {};
    #pragma unroll
    for (int ks = 0; ks < 2; ++ks) {
        v16bf A, Bv;
        #pragma unroll
        for (int e = 0; e < 16; ++e) {
            int kl = (e & 7) + ((e >> 3) << 4) + hi * 8;
            A[e]  = HU[(size_t)br * DD + ks * 32 + kl];
            Bv[e] = HR[(size_t)nc * DD + ks * 32 + kl];
        }
        acc = __builtin_amdgcn_wmma_f32_16x16x32_bf16(false, A, false, Bv,
                                                      (short)0, acc, false, false);
    }
    #pragma unroll
    for (int j = 0; j < 8; ++j) {
        int b = bt + j + 8 * hi;
        int n = nt + row;
        if (b < B && n < Nn) out[(size_t)b * Nn + n] = acc[j];
    }
}

// ---------------------------------------------------------------------------
static inline char* carve(char*& p, size_t bytes)
{
    char* r = p;
    p += (bytes + 255) & ~(size_t)255;
    return r;
}

extern "C" void kernel_launch(void* const* d_in, const int* in_sizes, int n_in,
                              void* d_out, int out_size, void* d_ws, size_t ws_size,
                              hipStream_t stream)
{
    const int*   user       = (const int*)  d_in[0];
    // d_in[1] = item_seq (unused by reference)
    const int*   ur_idx     = (const int*)  d_in[2];
    const int*   ru_idx     = (const int*)  d_in[3];
    const int*   rr_idx     = (const int*)  d_in[4];
    const int*   ri_idx     = (const int*)  d_in[5];
    const float* user_emb   = (const float*)d_in[6];
    const float* recipe_emb = (const float*)d_in[7];
    const float* ing_emb    = (const float*)d_in[8];
    const float* Ws_u  = (const float*)d_in[9];
    const float* Wn_u  = (const float*)d_in[10];
    const float* a_u   = (const float*)d_in[11];
    const float* Ws_ru = (const float*)d_in[12];
    const float* Wn_ru = (const float*)d_in[13];
    const float* a_ru  = (const float*)d_in[14];
    const float* Ws_rr = (const float*)d_in[15];
    const float* Wn_rr = (const float*)d_in[16];
    const float* a_rr  = (const float*)d_in[17];
    const float* Ws_ri = (const float*)d_in[18];
    const float* Wn_ri = (const float*)d_in[19];
    const float* a_ri  = (const float*)d_in[20];
    const float* W_u   = (const float*)d_in[21];
    const float* b_u   = (const float*)d_in[22];
    const float* W_r   = (const float*)d_in[23];
    const float* b_r   = (const float*)d_in[24];
    const float* W_rel = (const float*)d_in[25];
    const float* b_rel = (const float*)d_in[26];
    const float* q_rel = (const float*)d_in[27];

    const int B  = in_sizes[0];
    const int Nr = in_sizes[7] / DD;     // number of recipes

    // ---- workspace carve ----
    char* p = (char*)d_ws;
    float* uv   = (float*)carve(p, 4 * 2 * HH * DD * sizeof(float));
    bf16*  WnF0 = (bf16*)carve(p, 16384 * sizeof(bf16));
    bf16*  WnF1 = (bf16*)carve(p, 16384 * sizeof(bf16));
    bf16*  WnF2 = (bf16*)carve(p, 16384 * sizeof(bf16));
    bf16*  WnF3 = (bf16*)carve(p, 16384 * sizeof(bf16));
    bf16*  WuF  = (bf16*)carve(p, 16384 * sizeof(bf16));
    bf16*  WrF  = (bf16*)carve(p, 16384 * sizeof(bf16));
    float* HU   = (float*)carve(p, (size_t)B  * DD * sizeof(float));
    bf16*  HUbf = (bf16*) carve(p, (size_t)B  * DD * sizeof(bf16));
    float* hru  = (float*)carve(p, (size_t)Nr * DD * sizeof(float));
    float* hrr  = (float*)carve(p, (size_t)Nr * DD * sizeof(float));
    float* hri  = (float*)carve(p, (size_t)Nr * DD * sizeof(float));
    bf16*  HRbf = (bf16*) carve(p, (size_t)Nr * DD * sizeof(bf16));
    (void)ws_size; (void)n_in; (void)out_size;

    // ---- precompute folded attention vectors & bf16 weight fragments ----
    uv_kernel<<<8, 256, 0, stream>>>(Ws_u, Wn_u, a_u, Ws_ru, Wn_ru, a_ru,
                                     Ws_rr, Wn_rr, a_rr, Ws_ri, Wn_ri, a_ri, uv);
    wfrag_kernel<<<64, 256, 0, stream>>>(Wn_u,  WnF0, 256);
    wfrag_kernel<<<64, 256, 0, stream>>>(Wn_ru, WnF1, 256);
    wfrag_kernel<<<64, 256, 0, stream>>>(Wn_rr, WnF2, 256);
    wfrag_kernel<<<64, 256, 0, stream>>>(Wn_ri, WnF3, 256);
    wfrag_kernel<<<64, 256, 0, stream>>>(W_u,   WuF,  256);
    wfrag_kernel<<<64, 256, 0, stream>>>(W_r,   WrF,  256);

    // ---- node attention (4 relations) ----
    const int gbU = (B  + 15) / 16;
    const int gbR = (Nr + 15) / 16;
    node_attn_kernel<<<gbU, 32, 0, stream>>>(user_emb,   user,    recipe_emb, ur_idx,
                                             uv + 0 * 512, uv + 0 * 512 + 256,
                                             WnF0, WuF, b_u, HU,  HUbf, B);
    node_attn_kernel<<<gbR, 32, 0, stream>>>(recipe_emb, nullptr, user_emb,   ru_idx,
                                             uv + 1 * 512, uv + 1 * 512 + 256,
                                             WnF1, WrF, b_r, hru, (bf16*)nullptr, Nr);
    node_attn_kernel<<<gbR, 32, 0, stream>>>(recipe_emb, nullptr, recipe_emb, rr_idx,
                                             uv + 2 * 512, uv + 2 * 512 + 256,
                                             WnF2, WrF, b_r, hrr, (bf16*)nullptr, Nr);
    node_attn_kernel<<<gbR, 32, 0, stream>>>(recipe_emb, nullptr, ing_emb,    ri_idx,
                                             uv + 3 * 512, uv + 3 * 512 + 256,
                                             WnF3, WrF, b_r, hri, (bf16*)nullptr, Nr);

    // ---- relation-level attention ----
    relattn_kernel<<<(Nr + 255) / 256, 256, 0, stream>>>(hru, hrr, hri, W_rel, b_rel,
                                                         q_rel, HRbf, Nr);

    // ---- final score GEMM: pred[b,n] = H_U[b].H_R[n] ----
    dim3 g((Nr + 15) / 16, (B + 15) / 16);
    pred_kernel<<<g, 32, 0, stream>>>(HUbf, HRbf, (float*)d_out, B, Nr);
}
